// KoopmanOperator_70892730188393
// MI455X (gfx1250) — matrix operand order repaired and checked by reading, not compiled, not run on hardware
//
#include <hip/hip_runtime.h>

typedef __bf16 bf16_t;
typedef bf16_t v16bf __attribute__((ext_vector_type(16)));
typedef float  v8f   __attribute__((ext_vector_type(8)));
typedef int    v4i   __attribute__((ext_vector_type(4)));

#define DT_F   0.1f
#define BMAX_F 0.3f

// Keep region base offsets opaque so DS addressing becomes one base VGPR +
// 16-bit immediates (instead of per-load v_or on huge absolute constants).
// NOTE: must be a VGPR constraint — these offsets are threadIdx-derived
// (divergent), an "s" constraint causes an illegal VGPR->SGPR copy.
#define OPAQUE_U32(x) asm("" : "+v"(x))

// ---- LDS layout (bytes) ----
// B-operand tile image: 32(K)x16(N) bf16 = 1024B. Register image order:
//   chunk c in {0,1}: bytes [c*512 + lane*16 .. +15] = VGPRs 4c..4c+3 of `lane`
//   element (lane L, vgpr v, half h): n = L&15, k = (L>>4)*16 + 2v + h
// Tiles stored [outer][kb] so each GEMM's K-chain is contiguous (DS immediates).
#define ABUF   0          // A^T tiles: [j 0..15][kb 0..7]  = 131072 B
#define VBUF   131072     // Vcat tiles: [l 0..5][kb 0..7]  =  49152 B
#define UBUF   180224     // Ucat tiles: [j 0..15][kb 0..2] =  49152 B
#define AFBUF  229376     // a tile: 128 rows x 6 f32       =   3072 B
#define BOUNCE 232448     // 8 waves x 8192B (16 rows x 256 bf16, stride 512)
#define SMEM_BYTES 297984

__device__ __forceinline__ unsigned pack_bf16(float lo, float hi) {
  union { bf16_t b[2]; unsigned u; } p;
  p.b[0] = (bf16_t)lo; p.b[1] = (bf16_t)hi;
  return p.u;
}

// `base` already includes lane*16; `off` is a small constant (DS immediate).
__device__ __forceinline__ v16bf ldtile(const unsigned char* base, int off) {
  union { v4i q[2]; v16bf v; } u;
  u.q[0] = *(const v4i*)(base + off);
  u.q[1] = *(const v4i*)(base + off + 512);
  return u.v;
}

__global__ __launch_bounds__(256) void koopman_wmma_kernel(
    const float* __restrict__ zin, const float* __restrict__ ain,
    const float* __restrict__ Amat, const float* __restrict__ BU,
    const float* __restrict__ BV, const int* __restrict__ stepsPtr,
    float* __restrict__ out)
{
  __shared__ unsigned char smem[SMEM_BYTES];
  const int tid  = threadIdx.x;
  const int lane = tid & 31;
  const int wave = tid >> 5;
  const int n    = lane & 15;   // column within 16-wide tile
  const int half = lane >> 4;   // 0: rows v / K lo-half, 1: rows v+8 / K hi-half

  // ============ weight preload: f32 -> tanh/scale -> bf16, B-operand image ====
  // A^T tiles [j][kb]: B[k][n] = A[j*16+n][kb*32+k]
  for (int idx = tid; idx < 32768; idx += 256) {
    int tile = idx >> 8, w = idx & 255;
    int j = tile >> 3, kb = tile & 7;
    int c = w >> 7, rem = w & 127;
    int L = rem >> 2, vq = rem & 3, v = c * 4 + vq;
    int nn = L & 15;
    int k0 = (L >> 4) * 16 + 2 * v;
    float e0 = Amat[(j * 16 + nn) * 256 + kb * 32 + k0];
    float e1 = Amat[(j * 16 + nn) * 256 + kb * 32 + k0 + 1];
    *(unsigned*)(smem + ABUF + idx * 4) = pack_bf16(e0, e1);
  }
  // Vcat tiles [l][kb]: B[k][n] = tanh(BV[l][kb*32+k][n]) * BMAX
  for (int idx = tid; idx < 12288; idx += 256) {
    int tile = idx >> 8, w = idx & 255;
    int l = tile >> 3, kb = tile & 7;
    int c = w >> 7, rem = w & 127;
    int L = rem >> 2, vq = rem & 3, v = c * 4 + vq;
    int nn = L & 15;
    int k0 = kb * 32 + (L >> 4) * 16 + 2 * v;
    float e0 = tanhf(BV[l * 4096 + k0 * 16 + nn]) * BMAX_F;
    float e1 = tanhf(BV[l * 4096 + (k0 + 1) * 16 + nn]) * BMAX_F;
    *(unsigned*)(smem + VBUF + idx * 4) = pack_bf16(e0, e1);
  }
  // Ucat tiles [j][kb]: B[k][n] = tanh(BU[l][j*16+n][r]) * BMAX, l=k/16, r=k%16
  for (int idx = tid; idx < 12288; idx += 256) {
    int tile = idx >> 8, w = idx & 255;
    int j = tile / 3, kb = tile % 3;
    int c = w >> 7, rem = w & 127;
    int L = rem >> 2, vq = rem & 3, v = c * 4 + vq;
    int nn = L & 15;
    int kk0 = kb * 32 + (L >> 4) * 16 + 2 * v;
    int l0 = kk0 >> 4, r0 = kk0 & 15;
    int l1 = (kk0 + 1) >> 4, r1 = (kk0 + 1) & 15;
    float e0 = tanhf(BU[l0 * 4096 + (j * 16 + nn) * 16 + r0]) * BMAX_F;
    float e1 = tanhf(BU[l1 * 4096 + (j * 16 + nn) * 16 + r1]) * BMAX_F;
    *(unsigned*)(smem + UBUF + idx * 4) = pack_bf16(e0, e1);
  }
  // a tile (128 rows x 6, contiguous)
  {
    const float* asrc = ain + (size_t)blockIdx.x * 768;
    for (int idx = tid; idx < 768; idx += 256)
      *(float*)(smem + AFBUF + idx * 4) = asrc[idx];
  }
  __syncthreads();

  // ===== load this wave's 16x256 z tile into f32 C-layout accumulators =====
  const int rowBase = (blockIdx.x * 8 + wave) * 16;
  v8f zacc[16];
  #pragma unroll
  for (int j = 0; j < 16; ++j) {
    #pragma unroll
    for (int v = 0; v < 8; ++v) {
      int r = rowBase + v + half * 8;
      zacc[j][v] = __builtin_nontemporal_load(&zin[(size_t)r * 256 + j * 16 + n]);
    }
  }

  // Opaque region bases -> stable base registers + DS immediate offsets.
  unsigned aB = ABUF + lane * 16;
  unsigned vB = VBUF + lane * 16;
  unsigned uB = UBUF + lane * 16;
  unsigned afB = AFBUF + wave * 96 + half * 192;
  unsigned wbB = BOUNCE + wave * 8192;
  OPAQUE_U32(aB); OPAQUE_U32(vB); OPAQUE_U32(uB);
  OPAQUE_U32(afB); OPAQUE_U32(wbB);

  unsigned char* wb = smem + wbB;               // per-wave bounce, no barriers
  unsigned char* wbst = wb + n * 2;             // store base: + row*512 + tile*32
  const unsigned char* wbr = wb + n * 512 + half * 32;  // A-operand read base
  const unsigned char* afb = smem + afB;        // + (v*6+l)*4
  const int steps = *stepsPtr;

  for (int s = 0; s < steps; ++s) {
    // --- (1) z (f32 C-layout) -> row-major bf16 image -> A-operand tiles ---
    v16bf za[8];
    #pragma unroll
    for (int j = 0; j < 16; ++j) {
      #pragma unroll
      for (int v = 0; v < 8; ++v) {
        int row = v + half * 8;
        *(bf16_t*)(wbst + row * 512 + j * 32) = (bf16_t)zacc[j][v];
      }
    }
    asm volatile("s_wait_dscnt 0" ::: "memory");
    #pragma unroll
    for (int kb = 0; kb < 8; ++kb) {
      union { v4i q[2]; v16bf v; } u;
      u.q[0] = *(const v4i*)(wbr + kb * 64);
      u.q[1] = *(const v4i*)(wbr + kb * 64 + 16);
      za[kb] = u.v;
    }
    asm volatile("s_wait_dscnt 0" ::: "memory");

    // --- (2) proj = z @ Vcat (16x96), a-scale, bounce -> bf16 A tiles ---
    // Pairs of l -> two independent accumulator chains sharing za[kb].
    #pragma unroll
    for (int lp = 0; lp < 3; ++lp) {
      const unsigned char* V0 = smem + vB + (2 * lp) * 8192;
      const unsigned char* V1 = V0 + 8192;
      v8f acc0 = {0.f,0.f,0.f,0.f,0.f,0.f,0.f,0.f};
      v8f acc1 = {0.f,0.f,0.f,0.f,0.f,0.f,0.f,0.f};
      v16bf p0[2], p1[2];
      p0[0] = ldtile(V0, 0);    p1[0] = ldtile(V1, 0);
      p0[1] = ldtile(V0, 1024); p1[1] = ldtile(V1, 1024);
      #pragma unroll
      for (int kb = 0; kb < 8; ++kb) {
        v16bf c0 = p0[kb & 1], c1 = p1[kb & 1];
        if (kb + 2 < 8) {
          p0[kb & 1] = ldtile(V0, (kb + 2) * 1024);
          p1[kb & 1] = ldtile(V1, (kb + 2) * 1024);
        }
        acc0 = __builtin_amdgcn_wmma_f32_16x16x32_bf16(
            false, za[kb], false, c0, (short)0, acc0, false, false);
        acc1 = __builtin_amdgcn_wmma_f32_16x16x32_bf16(
            false, za[kb], false, c1, (short)0, acc1, false, false);
      }
      #pragma unroll
      for (int v = 0; v < 8; ++v) {
        int row = v + half * 8;
        float s0 = *(const float*)(afb + (v * 6 + 2 * lp) * 4);
        float s1 = *(const float*)(afb + (v * 6 + 2 * lp + 1) * 4);
        *(bf16_t*)(wbst + row * 512 + (2 * lp) * 32)     = (bf16_t)(acc0[v] * s0);
        *(bf16_t*)(wbst + row * 512 + (2 * lp + 1) * 32) = (bf16_t)(acc1[v] * s1);
      }
    }
    asm volatile("s_wait_dscnt 0" ::: "memory");
    v16bf pb[3];
    #pragma unroll
    for (int lp = 0; lp < 3; ++lp) {
      union { v4i q[2]; v16bf v; } u;
      u.q[0] = *(const v4i*)(wbr + lp * 64);
      u.q[1] = *(const v4i*)(wbr + lp * 64 + 16);
      pb[lp] = u.v;
    }
    asm volatile("s_wait_dscnt 0" ::: "memory");

    // --- (3) z_next = z + DT * ( z@A^T + scaled_proj@Ucat ) ---
    // Pairs of j: two chains share za/pb; unified 11-tile (8 A + 3 U) stream
    // with depth-2 prefetch.
    #pragma unroll
    for (int jp = 0; jp < 8; ++jp) {
      const unsigned char* A0 = smem + aB + (2 * jp) * 8192;
      const unsigned char* A1 = A0 + 8192;
      const unsigned char* U0 = smem + uB + (2 * jp) * 3072;
      const unsigned char* U1 = U0 + 3072;
      v8f acc0 = {0.f,0.f,0.f,0.f,0.f,0.f,0.f,0.f};
      v8f acc1 = {0.f,0.f,0.f,0.f,0.f,0.f,0.f,0.f};
      v16bf p0[2], p1[2];
      p0[0] = ldtile(A0, 0);    p1[0] = ldtile(A1, 0);
      p0[1] = ldtile(A0, 1024); p1[1] = ldtile(A1, 1024);
      #pragma unroll
      for (int t = 0; t < 11; ++t) {
        v16bf c0 = p0[t & 1], c1 = p1[t & 1];
        const int tn = t + 2;
        if (tn < 11) {
          p0[t & 1] = (tn < 8) ? ldtile(A0, tn * 1024) : ldtile(U0, (tn - 8) * 1024);
          p1[t & 1] = (tn < 8) ? ldtile(A1, tn * 1024) : ldtile(U1, (tn - 8) * 1024);
        }
        v16bf aop = (t < 8) ? za[t] : pb[t - 8];
        acc0 = __builtin_amdgcn_wmma_f32_16x16x32_bf16(
            false, aop, false, c0, (short)0, acc0, false, false);
        acc1 = __builtin_amdgcn_wmma_f32_16x16x32_bf16(
            false, aop, false, c1, (short)0, acc1, false, false);
      }
      #pragma unroll
      for (int v = 0; v < 8; ++v) {
        zacc[2 * jp][v]     = fmaf(DT_F, acc0[v], zacc[2 * jp][v]);
        zacc[2 * jp + 1][v] = fmaf(DT_F, acc1[v], zacc[2 * jp + 1][v]);
      }
    }
  }

  // ===== store result (streaming, non-temporal) =====
  #pragma unroll
  for (int j = 0; j < 16; ++j) {
    #pragma unroll
    for (int v = 0; v < 8; ++v) {
      int r = rowBase + v + half * 8;
      __builtin_nontemporal_store(zacc[j][v], &out[(size_t)r * 256 + j * 16 + n]);
    }
  }
}

extern "C" void kernel_launch(void* const* d_in, const int* in_sizes, int n_in,
                              void* d_out, int out_size, void* d_ws, size_t ws_size,
                              hipStream_t stream) {
  (void)n_in; (void)out_size; (void)d_ws; (void)ws_size;
  const float* z  = (const float*)d_in[0];
  const float* a  = (const float*)d_in[1];
  const float* A  = (const float*)d_in[2];
  const float* BU = (const float*)d_in[3];
  const float* BV = (const float*)d_in[4];
  const int* steps = (const int*)d_in[5];
  float* out = (float*)d_out;

  int N = in_sizes[0] / 256;      // rows (B*T)
  int blocks = N / 128;           // 8 waves x 16 rows per block
  hipLaunchKernelGGL(koopman_wmma_kernel, dim3(blocks), dim3(256), 0, stream,
                     z, a, A, BU, BV, steps, out);
}
